// ModelXLSTM_26121991094445
// MI455X (gfx1250) — matrix operand order repaired and checked by reading, not compiled, not run on hardware
//
#include <hip/hip_runtime.h>
#include <math.h>

typedef __attribute__((ext_vector_type(16))) _Float16 v16h;
typedef __attribute__((ext_vector_type(8)))  _Float16 v8h;
typedef __attribute__((ext_vector_type(4)))  _Float16 v4h;
typedef __attribute__((ext_vector_type(8)))  float    v8f;

#define Bz   8
#define Sq   1024
#define Dm   256
#define NHh  4
#define UPM  512        // mLSTM inner dim (2*D)
#define DHMh 128
#define DHSh 64
#define UPF  384
#define NTOK (Bz*Sq)    // 8192

#define CAT16(a, b) __builtin_shufflevector(a, b, 0,1,2,3,4,5,6,7,8,9,10,11,12,13,14,15)

// ---------------------------------------------------------------------------
// Embedding
// ---------------------------------------------------------------------------
__global__ __launch_bounds__(256) void k_embed(const int* __restrict__ tok,
                                               const int* __restrict__ tid,
                                               const float* __restrict__ te,
                                               const float* __restrict__ se,
                                               float* __restrict__ x) {
  int t = blockIdx.x, c = threadIdx.x;
  int tk = tok[t];
  int b = t >> 10;
  float v = se[tk * Dm + c] + te[tid[b] * Dm + c];
  x[(size_t)t * Dm + c] = (tk != 0) ? v : 0.0f;
}

// ---------------------------------------------------------------------------
// LayerNorm over D=256 (weight only)
// ---------------------------------------------------------------------------
__global__ __launch_bounds__(256) void k_layernorm(const float* __restrict__ x,
                                                   const float* __restrict__ w,
                                                   float* __restrict__ y) {
  __shared__ float red[256];
  int t = blockIdx.x, c = threadIdx.x;
  float v = x[(size_t)t * Dm + c];
  red[c] = v; __syncthreads();
  for (int s = 128; s > 0; s >>= 1) { if (c < s) red[c] += red[c + s]; __syncthreads(); }
  float mu = red[0] * (1.0f / Dm);
  __syncthreads();
  float d = v - mu;
  red[c] = d * d; __syncthreads();
  for (int s = 128; s > 0; s >>= 1) { if (c < s) red[c] += red[c + s]; __syncthreads(); }
  float var = red[0] * (1.0f / Dm);
  y[(size_t)t * Dm + c] = d * rsqrtf(var + 1e-5f) * w[c];
}

// ---------------------------------------------------------------------------
// WMMA GEMM: C[M,N] (+=) A[M,K] * B[K,N].  A row-major (lda), B either
// [K,N] (btrans=0, element (k,n)=B[k*ldb+n]) or [N,K] (btrans=1).
// f16 via LDS (B staged transposed so fragments are contiguous), fp32 acc.
// M%64==0, N%64==0, K%32==0.  128 threads = 4 waves, 64x64 tile.
// ---------------------------------------------------------------------------
__global__ __launch_bounds__(128) void k_gemm(const float* __restrict__ A, int lda,
                                              const float* __restrict__ B, int ldb, int btrans,
                                              float* __restrict__ C, int ldc,
                                              const float* __restrict__ bias, int beta,
                                              int M, int N, int K) {
  __shared__ __align__(16) _Float16 sA [64][40];   // [m][k]   row pitch 80B
  __shared__ __align__(16) _Float16 sBt[64][40];   // [n][k]   row pitch 80B
  int tm = blockIdx.x * 64, tn = blockIdx.y * 64;
  int tid3 = threadIdx.x, wave = tid3 >> 5, lane = tid3 & 31;
  int Ml = lane & 15, grp = lane >> 4;
  v8f acc[4] = {};
  for (int k0 = 0; k0 < K; k0 += 32) {
    __syncthreads();
    if (k0 + 32 < K)  // CDNA5 global_prefetch_b8 for next A tile
      __builtin_prefetch(&A[(size_t)(tm + (tid3 >> 1)) * lda + k0 + 32], 0, 1);
    // A: 64x32, float4 global loads -> v4h LDS stores
#pragma unroll
    for (int j = 0; j < 4; ++j) {
      int q = tid3 + j * 128;            // quad id over 64*8
      int r = q >> 3, c4 = (q & 7) * 4;
      float4 f = *(const float4*)&A[(size_t)(tm + r) * lda + k0 + c4];
      v4h h4; h4[0] = (_Float16)f.x; h4[1] = (_Float16)f.y;
      h4[2] = (_Float16)f.z; h4[3] = (_Float16)f.w;
      *(v4h*)&sA[r][c4] = h4;
    }
    // B: 32x64 staged transposed into sBt[n][k]
#pragma unroll
    for (int j = 0; j < 4; ++j) {
      int q = tid3 + j * 128;            // quad id over 64n * 8kq
      int n = q >> 3, k4 = (q & 7) * 4;
      v4h h4;
      if (btrans) {
        float4 f = *(const float4*)&B[(size_t)(tn + n) * ldb + k0 + k4];
        h4[0] = (_Float16)f.x; h4[1] = (_Float16)f.y;
        h4[2] = (_Float16)f.z; h4[3] = (_Float16)f.w;
      } else {
#pragma unroll
        for (int i = 0; i < 4; ++i)
          h4[i] = (_Float16)B[(size_t)(k0 + k4 + i) * ldb + tn + n];
      }
      *(v4h*)&sBt[n][k4] = h4;
    }
    __syncthreads();
    // A fragment: two contiguous 8-half runs per ISA 16x32 layout
    int row = wave * 16 + Ml;
    v8h a0 = *(const v8h*)&sA[row][grp << 3];
    v8h a1 = *(const v8h*)&sA[row][16 + (grp << 3)];
    v16h af = CAT16(a0, a1);
#pragma unroll
    for (int sub = 0; sub < 4; ++sub) {
      int col = sub * 16 + Ml;
      v8h b0 = *(const v8h*)&sBt[col][grp << 3];
      v8h b1 = *(const v8h*)&sBt[col][16 + (grp << 3)];
      v16h bf = CAT16(b0, b1);
      acc[sub] = __builtin_amdgcn_wmma_f32_16x16x32_f16(
          false, af, false, bf, (short)0, acc[sub], false, false);
    }
  }
#pragma unroll
  for (int sub = 0; sub < 4; ++sub) {
    int col = tn + sub * 16 + Ml;
    float bv = bias ? bias[sub * 16 + Ml + tn] : 0.0f;  // bias is call-local base
#pragma unroll
    for (int r = 0; r < 8; ++r) {
      int rowg = tm + wave * 16 + r + (grp << 3);
      size_t idx = (size_t)rowg * ldc + col;
      float v = acc[sub][r] + bv;
      C[idx] = beta ? (C[idx] + v) : v;
    }
  }
}

// ---------------------------------------------------------------------------
// Depthwise causal conv1d (K=4) + SiLU
// ---------------------------------------------------------------------------
__global__ __launch_bounds__(256) void k_conv_silu(const float* __restrict__ in, int lda,
                                                   const float* __restrict__ w,
                                                   const float* __restrict__ bias,
                                                   float* __restrict__ out, int C, int total) {
  int idx = blockIdx.x * 256 + threadIdx.x;
  if (idx >= total) return;
  int t = idx / C, c = idx - t * C;
  int s = t & (Sq - 1);
  float y = bias[c];
#pragma unroll
  for (int k = 0; k < 4; ++k) {
    int sp = s - 3 + k;
    if (sp >= 0) y += in[(size_t)(t - 3 + k) * lda + c] * w[k * C + c];
  }
  out[idx] = y / (1.0f + __expf(-y));
}

// ---------------------------------------------------------------------------
// Block-diagonal headwise projection (blocksize 4)
// ---------------------------------------------------------------------------
__global__ __launch_bounds__(256) void k_blockdiag(const float* __restrict__ in, int lda,
                                                   const float* __restrict__ W,
                                                   float* __restrict__ out) {
  int idx = blockIdx.x * 256 + threadIdx.x;   // t*512 + c
  int t = idx >> 9, c = idx & 511;
  int n = c >> 2, o = c & 3;
  const float* wb = W + n * 16 + o * 4;
  const float* ib = in + (size_t)t * lda + n * 4;
  out[(size_t)t * UPM + c] = ib[0]*wb[0] + ib[1]*wb[1] + ib[2]*wb[2] + ib[3]*wb[3];
}

// ---------------------------------------------------------------------------
// mLSTM i/f gate projections: one thread per token, all 4 heads at once so
// W rows load as float4.  Output [B,NH,S].
// ---------------------------------------------------------------------------
__global__ __launch_bounds__(256) void k_mlstm_gate(const float* __restrict__ q,
                                                    const float* __restrict__ k,
                                                    const float* __restrict__ v,
                                                    const float* __restrict__ Wig,
                                                    const float* __restrict__ big,
                                                    const float* __restrict__ Wfg,
                                                    const float* __restrict__ bfg,
                                                    float* __restrict__ ig,
                                                    float* __restrict__ fg) {
  int t = blockIdx.x * 256 + threadIdx.x;
  if (t >= NTOK) return;
  const float* qr = q + (size_t)t * UPM;
  const float* kr = k + (size_t)t * UPM;
  const float* vr = v + (size_t)t * UPM;
  float4 si = *(const float4*)big;
  float4 sf = *(const float4*)bfg;
  for (int c = 0; c < UPM; ++c) {
    float qc = qr[c], kc = kr[c], vc = vr[c];
    float4 wiq = *(const float4*)&Wig[(size_t)c * 4];
    float4 wik = *(const float4*)&Wig[(size_t)(UPM + c) * 4];
    float4 wiv = *(const float4*)&Wig[(size_t)(2 * UPM + c) * 4];
    float4 wfq = *(const float4*)&Wfg[(size_t)c * 4];
    float4 wfk = *(const float4*)&Wfg[(size_t)(UPM + c) * 4];
    float4 wfv = *(const float4*)&Wfg[(size_t)(2 * UPM + c) * 4];
    si.x += qc * wiq.x + kc * wik.x + vc * wiv.x;
    si.y += qc * wiq.y + kc * wik.y + vc * wiv.y;
    si.z += qc * wiq.z + kc * wik.z + vc * wiv.z;
    si.w += qc * wiq.w + kc * wik.w + vc * wiv.w;
    sf.x += qc * wfq.x + kc * wfk.x + vc * wfv.x;
    sf.y += qc * wfq.y + kc * wfk.y + vc * wfv.y;
    sf.z += qc * wfq.z + kc * wfk.z + vc * wfv.z;
    sf.w += qc * wfq.w + kc * wfk.w + vc * wfv.w;
  }
  int b = t >> 10, s = t & (Sq - 1);
  size_t o = ((size_t)b * NHh << 10) + s;
  ig[o]          = si.x; fg[o]          = sf.x;
  ig[o + 1024]   = si.y; fg[o + 1024]   = sf.y;
  ig[o + 2048]   = si.z; fg[o + 2048]   = sf.z;
  ig[o + 3072]   = si.w; fg[o + 3072]   = sf.w;
}

// cumsum of log-sigmoid(fg) along S, per (b,h)
__global__ void k_lfc(const float* __restrict__ fg, float* __restrict__ lfc) {
  int bh = blockIdx.x;
  const float* f = fg + (size_t)bh * Sq;
  float* o = lfc + (size_t)bh * Sq;
  float acc = 0.0f;
  for (int s = 0; s < Sq; ++s) {
    float x = f[s];
    acc += fminf(x, 0.0f) - log1pf(__expf(-fabsf(x)));
    o[s] = acc;
  }
}

// ---------------------------------------------------------------------------
// Flash-style stabilized mLSTM matrix memory (one wave per (b,h,16 rows)).
// norm = max(|sum C|, exp(-mmax)); h = (C/(norm+1e-6)) @ V  (as reference).
// ---------------------------------------------------------------------------
#define NEGBIG (-1e30f)
__global__ __launch_bounds__(32) void k_mlstm_attn(const float* __restrict__ Q,
                                                   const float* __restrict__ Km,
                                                   const float* __restrict__ V,
                                                   const float* __restrict__ lfc,
                                                   const float* __restrict__ ig,
                                                   float* __restrict__ H) {
  const int tb = blockIdx.x, hh = blockIdx.y, bb = blockIdx.z;
  const int lane = threadIdx.x;
  const int t0 = tb * 16;
  const size_t rowbase = (size_t)bb * Sq * UPM + hh * DHMh;
  const int Ml = lane & 15, grp = lane >> 4;

  __shared__ __align__(16) _Float16 sK [16][DHMh + 8];  // [s][d], pitch 272B
  __shared__ __align__(16) _Float16 sVt[DHMh][16];      // [d][s], pitch 32B
  __shared__ __align__(16) _Float16 sP [16][24];        // [t][s], pitch 48B

  // Q fragments via float4 loads, pre-scaled by 1/sqrt(DHM)
  v16h qa[4];
  {
    const float* qrow = Q + rowbase + (size_t)(t0 + Ml) * UPM;
#pragma unroll
    for (int c = 0; c < 4; ++c) {
      int o0 = c * 32 + (grp << 3);
      float4 f0 = *(const float4*)(qrow + o0);
      float4 f1 = *(const float4*)(qrow + o0 + 4);
      float4 f2 = *(const float4*)(qrow + o0 + 16);
      float4 f3 = *(const float4*)(qrow + o0 + 20);
      const float sc16 = 0.08838834764831845f;
      qa[c][0] = (_Float16)(f0.x*sc16); qa[c][1] = (_Float16)(f0.y*sc16);
      qa[c][2] = (_Float16)(f0.z*sc16); qa[c][3] = (_Float16)(f0.w*sc16);
      qa[c][4] = (_Float16)(f1.x*sc16); qa[c][5] = (_Float16)(f1.y*sc16);
      qa[c][6] = (_Float16)(f1.z*sc16); qa[c][7] = (_Float16)(f1.w*sc16);
      qa[c][8] = (_Float16)(f2.x*sc16); qa[c][9] = (_Float16)(f2.y*sc16);
      qa[c][10] = (_Float16)(f2.z*sc16); qa[c][11] = (_Float16)(f2.w*sc16);
      qa[c][12] = (_Float16)(f3.x*sc16); qa[c][13] = (_Float16)(f3.y*sc16);
      qa[c][14] = (_Float16)(f3.z*sc16); qa[c][15] = (_Float16)(f3.w*sc16);
    }
  }

  const float* lfp = lfc + ((size_t)bb * NHh + hh) * Sq;
  const float* igp = ig  + ((size_t)bb * NHh + hh) * Sq;
  float lft[8];
#pragma unroll
  for (int r = 0; r < 8; ++r) lft[r] = lfp[t0 + r + (grp << 3)];

  v8f acc[8] = {};
  float m_run[8], s_run[8];
#pragma unroll
  for (int r = 0; r < 8; ++r) { m_run[r] = NEGBIG; s_run[r] = 0.0f; }

  for (int s0 = 0; s0 <= t0; s0 += 16) {
    __syncthreads();
    // stage K row-major (v4h stores) and V transposed
#pragma unroll
    for (int j = 0; j < 16; ++j) {
      int qd = lane + j * 32;            // quad over 16*32
      int rr = qd >> 5, cc4 = (qd & 31) * 4;
      float4 f = *(const float4*)&Km[rowbase + (size_t)(s0 + rr) * UPM + cc4];
      v4h h4; h4[0] = (_Float16)f.x; h4[1] = (_Float16)f.y;
      h4[2] = (_Float16)f.z; h4[3] = (_Float16)f.w;
      *(v4h*)&sK[rr][cc4] = h4;
      float4 g = *(const float4*)&V[rowbase + (size_t)(s0 + rr) * UPM + cc4];
      sVt[cc4 + 0][rr] = (_Float16)g.x;
      sVt[cc4 + 1][rr] = (_Float16)g.y;
      sVt[cc4 + 2][rr] = (_Float16)g.z;
      sVt[cc4 + 3][rr] = (_Float16)g.w;
    }
    __syncthreads();

    // S = q . k^T (16x16 tile): B fragment is contiguous along features
    v8f sc = {};
#pragma unroll
    for (int c = 0; c < 4; ++c) {
      int o0 = c * 32 + (grp << 3);
      v8h b0 = *(const v8h*)&sK[Ml][o0];
      v8h b1 = *(const v8h*)&sK[Ml][o0 + 16];
      v16h kb = CAT16(b0, b1);
      sc = __builtin_amdgcn_wmma_f32_16x16x32_f16(false, qa[c], false, kb,
                                                  (short)0, sc, false, false);
    }

    int s_idx = s0 + Ml;
    float lf_s = lfp[s_idx], ig_s = igp[s_idx];
    float alpha[8];
#pragma unroll
    for (int r = 0; r < 8; ++r) {
      int t_idx = t0 + r + (grp << 3);
      float dm = (s_idx <= t_idx) ? (lft[r] - lf_s + ig_s) : NEGBIG;
      float tmax = dm;
#pragma unroll
      for (int m2 = 1; m2 < 16; m2 <<= 1) tmax = fmaxf(tmax, __shfl_xor(tmax, m2, 32));
      float mnew = fmaxf(m_run[r], tmax);
      alpha[r] = __expf(m_run[r] - mnew);
      float p = sc[r] * __expf(dm - mnew);
      float ps = p;
#pragma unroll
      for (int m2 = 1; m2 < 16; m2 <<= 1) ps += __shfl_xor(ps, m2, 32);
      s_run[r] = s_run[r] * alpha[r] + ps;
      m_run[r] = mnew;
      sP[r + (grp << 3)][Ml] = (_Float16)p;
    }
#pragma unroll
    for (int sub = 0; sub < 8; ++sub)
#pragma unroll
      for (int r = 0; r < 8; ++r) acc[sub][r] *= alpha[r];
    __syncthreads();

    // P (16x16, K padded to 32 with zeros in A operand) @ V (16x128).
    // A zeros cover K=16..31 across the striped layout, so the K>=16 rows of
    // the B operand never contribute -- no masking needed on vb.
    v16h pa;
    {
      v8h plo = *(const v8h*)&sP[Ml][grp << 3];
#pragma unroll
      for (int e = 0; e < 8; ++e)  pa[e] = plo[e];
#pragma unroll
      for (int e = 8; e < 16; ++e) pa[e] = (_Float16)0.0f;
    }
#pragma unroll
    for (int sub = 0; sub < 8; ++sub) {
      int col = sub * 16 + Ml;
      v8h v0 = *(const v8h*)&sVt[col][grp << 3];
      v8h v1 = *(const v8h*)&sVt[col][8];
      v16h vb = CAT16(v0, v1);
      acc[sub] = __builtin_amdgcn_wmma_f32_16x16x32_f16(false, pa, false, vb,
                                                        (short)0, acc[sub], false, false);
    }
  }

  float inv[8];
#pragma unroll
  for (int r = 0; r < 8; ++r) {
    float norm = fmaxf(fabsf(s_run[r]), __expf(-m_run[r])) + 1e-6f;
    inv[r] = 1.0f / norm;
  }
#pragma unroll
  for (int sub = 0; sub < 8; ++sub)
#pragma unroll
    for (int r = 0; r < 8; ++r) {
      int row = t0 + r + (grp << 3);
      H[rowbase + (size_t)row * UPM + sub * 16 + Ml] = acc[sub][r] * inv[r];
    }
}

// ---------------------------------------------------------------------------
// mLSTM combine: per-head norm(H) + skip*xc, gated by silu(z)
// ---------------------------------------------------------------------------
__global__ __launch_bounds__(128) void k_mlstm_combine(const float* __restrict__ H,
                                                       const float* __restrict__ xc,
                                                       const float* __restrict__ up,
                                                       const float* __restrict__ mnw,
                                                       const float* __restrict__ skp,
                                                       float* __restrict__ out) {
  __shared__ float red[128];
  int t = blockIdx.x >> 2, h = blockIdx.x & 3;
  int c = threadIdx.x, ch = h * DHMh + c;
  float v = H[(size_t)t * UPM + ch];
  red[c] = v; __syncthreads();
  for (int s = 64; s > 0; s >>= 1) { if (c < s) red[c] += red[c + s]; __syncthreads(); }
  float mu = red[0] * (1.0f / DHMh);
  __syncthreads();
  float d = v - mu;
  red[c] = d * d; __syncthreads();
  for (int s = 64; s > 0; s >>= 1) { if (c < s) red[c] += red[c + s]; __syncthreads(); }
  float var = red[0] * (1.0f / DHMh);
  float hn = d * rsqrtf(var + 1e-5f) * mnw[ch] + skp[ch] * xc[(size_t)t * UPM + ch];
  float z = up[(size_t)t * (2 * UPM) + UPM + ch];
  out[(size_t)t * UPM + ch] = hn * (z / (1.0f + __expf(-z)));
}

// ---------------------------------------------------------------------------
// sLSTM recurrent scan: one workgroup per batch, thread = (head, j)
// ---------------------------------------------------------------------------
__global__ __launch_bounds__(256) void k_slstm_scan(const float* __restrict__ G,
                                                    const float* __restrict__ R,
                                                    float* __restrict__ HS) {
  int b = blockIdx.x, t = threadIdx.x;
  int n = t >> 6, j = t & 63;
  __shared__ __align__(16) float hs_s[256];
  const float* Ri = R + (0 * 4 + n) * 4096 + j * 64;
  const float* Rf = R + (1 * 4 + n) * 4096 + j * 64;
  const float* Rz = R + (2 * 4 + n) * 4096 + j * 64;
  const float* Ro = R + (3 * 4 + n) * 4096 + j * 64;
  float cst = 0.0f, nst = 0.0f, mst = 0.0f;
  hs_s[t] = 0.0f; __syncthreads();
  const float* g = G + (size_t)b * Sq * 1024;
  for (int s = 0; s < Sq; ++s) {
    const float* hb = hs_s + n * 64;
    float di = 0, df = 0, dz = 0, dop = 0;
    for (int i = 0; i < 64; i += 4) {
      float4 hv = *(const float4*)&hb[i];
      float4 ri = *(const float4*)&Ri[i];
      float4 rf = *(const float4*)&Rf[i];
      float4 rz = *(const float4*)&Rz[i];
      float4 ro = *(const float4*)&Ro[i];
      di += hv.x*ri.x + hv.y*ri.y + hv.z*ri.z + hv.w*ri.w;
      df += hv.x*rf.x + hv.y*rf.y + hv.z*rf.z + hv.w*rf.w;
      dz += hv.x*rz.x + hv.y*rz.y + hv.z*rz.z + hv.w*rz.w;
      dop += hv.x*ro.x + hv.y*ro.y + hv.z*ro.z + hv.w*ro.w;
    }
    const float* gs = g + (size_t)s * 1024;
    float iraw = gs[0 * 256 + t] + di;
    float fraw = gs[1 * 256 + t] + df;
    float zt = tanhf(gs[2 * 256 + t] + dz);
    float og = gs[3 * 256 + t] + dop;
    float ot = 1.0f / (1.0f + __expf(-og));
    float lfm = mst + fminf(fraw, 0.0f) - log1pf(__expf(-fabsf(fraw)));
    float mnew = fmaxf(iraw, lfm);
    float it = __expf(iraw - mnew);
    float ft = __expf(lfm - mnew);
    cst = ft * cst + it * zt;
    nst = ft * nst + it;
    float hnew = ot * cst / fmaxf(nst, 1e-6f);
    mst = mnew;
    __syncthreads();
    hs_s[t] = hnew;
    HS[((size_t)b * Sq + s) * Dm + t] = hnew;
    __syncthreads();
  }
}

// per-head (group of 64) norm then residual add into x
__global__ __launch_bounds__(256) void k_headnorm_add(const float* __restrict__ hin,
                                                      const float* __restrict__ w,
                                                      float* __restrict__ x) {
  __shared__ float red[256];
  int t = blockIdx.x, c = threadIdx.x, gp = c >> 6, l = c & 63;
  float v = hin[(size_t)t * Dm + c];
  red[c] = v; __syncthreads();
  for (int s = 32; s > 0; s >>= 1) { if (l < s) red[c] += red[c + s]; __syncthreads(); }
  float mu = red[gp * 64] * (1.0f / 64.0f);
  __syncthreads();
  float d = v - mu;
  red[c] = d * d; __syncthreads();
  for (int s = 32; s > 0; s >>= 1) { if (l < s) red[c] += red[c + s]; __syncthreads(); }
  float var = red[gp * 64] * (1.0f / 64.0f);
  x[(size_t)t * Dm + c] += d * rsqrtf(var + 1e-5f) * w[c];
}

// gated FFN activation: gelu_tanh(a) * g
__global__ __launch_bounds__(256) void k_ffn_act(const float* __restrict__ up,
                                                 float* __restrict__ out, int total) {
  int idx = blockIdx.x * 256 + threadIdx.x;
  if (idx >= total) return;
  int t = idx / UPF, c = idx - t * UPF;
  float a = up[(size_t)t * (2 * UPF) + c];
  float gv = up[(size_t)t * (2 * UPF) + UPF + c];
  float inner = 0.7978845608028654f * (a + 0.044715f * a * a * a);
  out[idx] = 0.5f * a * (1.0f + tanhf(inner)) * gv;
}

__global__ __launch_bounds__(256) void k_gather_last(const float* __restrict__ xln,
                                                     const int* __restrict__ slen,
                                                     float* __restrict__ last) {
  int b = blockIdx.x, c = threadIdx.x;
  int s = slen[b] - 1;
  last[b * Dm + c] = xln[((size_t)b * Sq + s) * Dm + c];
}

__global__ __launch_bounds__(256) void k_head1(const float* __restrict__ last,
                                               const float* __restrict__ W,
                                               const float* __restrict__ bia,
                                               float* __restrict__ h1) {
  int b = blockIdx.x, j = threadIdx.x;
  float acc = bia[j];
  for (int c = 0; c < Dm; ++c) acc += last[b * Dm + c] * W[c * Dm + j];
  h1[b * Dm + j] = fmaxf(acc, 0.0f);
}

__global__ void k_head2(const float* __restrict__ h1, const float* __restrict__ W2,
                        const float* __restrict__ b2, float* __restrict__ out) {
  int b = threadIdx.x;
  if (b >= Bz) return;
  float acc = b2[0];
  for (int c = 0; c < Dm; ++c) acc += h1[b * Dm + c] * W2[c];
  out[b] = acc;
}

// ---------------------------------------------------------------------------
extern "C" void kernel_launch(void* const* d_in, const int* in_sizes, int n_in,
                              void* d_out, int out_size, void* d_ws, size_t ws_size,
                              hipStream_t stream) {
  (void)in_sizes; (void)n_in; (void)out_size; (void)ws_size;
  const int* tok  = (const int*)d_in[0];
  const int* tid  = (const int*)d_in[1];
  const int* slen = (const int*)d_in[2];
  const float* te = (const float*)d_in[3];
  const float* se = (const float*)d_in[4];

  char* ws = (char*)d_ws;
  const size_t MBy = 1024ull * 1024ull;
  float* X    = (float*)(ws + 0);           //  8 MB  residual
  float* XLN  = (float*)(ws + 8 * MBy);     //  8 MB
  float* UPb  = (float*)(ws + 16 * MBy);    // 32 MB  (mLSTM up / sLSTM gates / FFN up)
  float* XCb  = (float*)(ws + 48 * MBy);    // 16 MB  (conv out / FFN act)
  float* Qb   = (float*)(ws + 64 * MBy);    // 16 MB
  float* Kb   = (float*)(ws + 80 * MBy);    // 16 MB
  float* Vb   = (float*)(ws + 96 * MBy);    // 16 MB
  float* Hb   = (float*)(ws + 112 * MBy);   // 16 MB  (attn out / scan out)
  float* IGb  = (float*)(ws + 128 * MBy);
  float* FGb  = IGb + Bz * NHh * Sq;
  float* LFC  = FGb + Bz * NHh * Sq;
  float* LAST = LFC + Bz * NHh * Sq;
  float* H1b  = LAST + Bz * Dm;

  auto gemm = [&](const float* A, int lda, const float* B, int ldb, int btr,
                  float* C, int ldc, const float* bias, int beta,
                  int M, int N, int K) {
    dim3 g(M / 64, N / 64);
    k_gemm<<<g, 128, 0, stream>>>(A, lda, B, ldb, btr, C, ldc, bias, beta, M, N, K);
  };

  k_embed<<<NTOK, 256, 0, stream>>>(tok, tid, te, se, X);

  int idx = 5;
  for (int bi = 0; bi < 4; ++bi) {
    bool is_s = (bi == 1 || bi == 3);
    if (!is_s) {
      const float* ln_w  = (const float*)d_in[idx + 0];
      const float* W_upm = (const float*)d_in[idx + 1];
      const float* cw    = (const float*)d_in[idx + 2];
      const float* cb    = (const float*)d_in[idx + 3];
      const float* Wq    = (const float*)d_in[idx + 4];
      const float* Wk    = (const float*)d_in[idx + 5];
      const float* Wv    = (const float*)d_in[idx + 6];
      const float* Wig   = (const float*)d_in[idx + 7];
      const float* big   = (const float*)d_in[idx + 8];
      const float* Wfg   = (const float*)d_in[idx + 9];
      const float* bfg   = (const float*)d_in[idx + 10];
      const float* mnw   = (const float*)d_in[idx + 11];
      const float* skp   = (const float*)d_in[idx + 12];
      const float* Wdnm  = (const float*)d_in[idx + 13];
      idx += 14;

      k_layernorm<<<NTOK, 256, 0, stream>>>(X, ln_w, XLN);
      gemm(XLN, Dm, W_upm, 2 * UPM, 0, UPb, 2 * UPM, nullptr, 0, NTOK, 2 * UPM, Dm);
      k_conv_silu<<<(NTOK * UPM) / 256, 256, 0, stream>>>(UPb, 2 * UPM, cw, cb, XCb,
                                                          UPM, NTOK * UPM);
      k_blockdiag<<<(NTOK * UPM) / 256, 256, 0, stream>>>(XCb, UPM, Wq, Qb);
      k_blockdiag<<<(NTOK * UPM) / 256, 256, 0, stream>>>(XCb, UPM, Wk, Kb);
      k_blockdiag<<<(NTOK * UPM) / 256, 256, 0, stream>>>(UPb, 2 * UPM, Wv, Vb);
      k_mlstm_gate<<<NTOK / 256, 256, 0, stream>>>(Qb, Kb, Vb, Wig, big,
                                                   Wfg, bfg, IGb, FGb);
      k_lfc<<<Bz * NHh, 1, 0, stream>>>(FGb, LFC);
      dim3 ga(Sq / 16, NHh, Bz);
      k_mlstm_attn<<<ga, 32, 0, stream>>>(Qb, Kb, Vb, LFC, IGb, Hb);
      k_mlstm_combine<<<NTOK * NHh, 128, 0, stream>>>(Hb, XCb, UPb, mnw, skp, Qb);
      gemm(Qb, UPM, Wdnm, Dm, 0, X, Dm, nullptr, 1, NTOK, Dm, UPM);
    } else {
      const float* ln_w = (const float*)d_in[idx + 0];
      const float* cw   = (const float*)d_in[idx + 1];
      const float* cb   = (const float*)d_in[idx + 2];
      const float* Wi   = (const float*)d_in[idx + 3];
      const float* Wf   = (const float*)d_in[idx + 4];
      const float* Wz   = (const float*)d_in[idx + 5];
      const float* Wo   = (const float*)d_in[idx + 6];
      const float* Rr   = (const float*)d_in[idx + 7];
      const float* bg   = (const float*)d_in[idx + 8];
      const float* gnw  = (const float*)d_in[idx + 9];
      const float* ln2w = (const float*)d_in[idx + 10];
      const float* Wup  = (const float*)d_in[idx + 11];
      const float* Wdn  = (const float*)d_in[idx + 12];
      idx += 13;

      k_layernorm<<<NTOK, 256, 0, stream>>>(X, ln_w, XLN);
      k_conv_silu<<<(NTOK * Dm) / 256, 256, 0, stream>>>(XLN, Dm, cw, cb, XCb,
                                                         Dm, NTOK * Dm);
      const float* Wg[4] = {Wi, Wf, Wz, Wo};
      for (int g2 = 0; g2 < 4; ++g2) {
        const float* Ain = (g2 < 2) ? XCb : XLN;
        for (int n = 0; n < 4; ++n) {
          gemm(Ain + n * 64, Dm, Wg[g2] + n * 4096, 64, 1,
               UPb + g2 * 256 + n * 64, 1024, bg + g2 * 256 + n * 64, 0,
               NTOK, 64, 64);
        }
      }
      k_slstm_scan<<<Bz, 256, 0, stream>>>(UPb, Rr, Hb);
      k_headnorm_add<<<NTOK, 256, 0, stream>>>(Hb, gnw, X);
      // gated FFN
      k_layernorm<<<NTOK, 256, 0, stream>>>(X, ln2w, XLN);
      gemm(XLN, Dm, Wup, 2 * UPF, 0, UPb, 2 * UPF, nullptr, 0, NTOK, 2 * UPF, Dm);
      k_ffn_act<<<(NTOK * UPF) / 256, 256, 0, stream>>>(UPb, XCb, NTOK * UPF);
      gemm(XCb, UPF, Wdn, Dm, 0, X, Dm, nullptr, 1, NTOK, Dm, UPF);
    }
  }

  const float* postw = (const float*)d_in[idx + 0];
  const float* Wp1   = (const float*)d_in[idx + 1];
  const float* bp1   = (const float*)d_in[idx + 2];
  const float* Wp2   = (const float*)d_in[idx + 3];
  const float* bp2   = (const float*)d_in[idx + 4];
  k_layernorm<<<NTOK, 256, 0, stream>>>(X, postw, XLN);
  k_gather_last<<<Bz, 256, 0, stream>>>(XLN, slen, LAST);
  k_head1<<<Bz, 256, 0, stream>>>(LAST, Wp1, bp1, H1b);
  k_head2<<<1, 32, 0, stream>>>(H1b, Wp2, bp2, (float*)d_out);
}
